// ProteinDenoiser_24824910971361
// MI455X (gfx1250) — matrix operand order repaired and checked by reading, not compiled, not run on hardware
//
#include <hip/hip_runtime.h>
#include <math.h>

#define B_    8
#define N_    1024
#define XD_   20
#define ED_   5
#define H_    384
#define NH_   16
#define DH_   24
#define MLPH_ 1536
#define FREQ_ 256
#define KNN_  30
#define TE_   (B_*N_*KNN_)

typedef __attribute__((ext_vector_type(16))) _Float16 v16h;
typedef __attribute__((ext_vector_type(8)))  _Float16 v8h;
typedef __attribute__((ext_vector_type(4)))  _Float16 v4h;
typedef __attribute__((ext_vector_type(8)))  float    v8f;

// ---------------------------------------------------------------------------
// WMMA fragment loaders (CDNA5 16-bit layouts, CDNA5 ISA §7.12.2)
// A 16x32 f16: lane(0-15) row=lid holds K{0..7,16..23}; lane(16-31) K{8..15,24..31}
//   -> per lane: two contiguous 8-half (16B) runs at +hi*8 and +16+hi*8.
// B 32x16 f16 from LDS stored TRANSPOSED as BsT[col][k]:
//   lane holds column col0+lid, K{hi*8..+7, 16+hi*8..+7} (same as A).
// ld (in halves) must be a multiple of 8 so the b128 loads stay 16B-aligned.
// ---------------------------------------------------------------------------
__device__ __forceinline__ v16h concat8(v8h a, v8h b) {
  return __builtin_shufflevector(a, b, 0, 1, 2, 3, 4, 5, 6, 7, 8, 9, 10, 11,
                                 12, 13, 14, 15);
}

__device__ __forceinline__ v16h load_a_frag(const _Float16* As, int ld) {
  const int lane = threadIdx.x & 31;
  const int lid  = lane & 15;
  const int hi   = lane >> 4;
  const _Float16* r = As + lid * ld + (hi << 3);
  v8h lo = *(const v8h*)(r);        // K = hi*8 .. hi*8+7
  v8h hs = *(const v8h*)(r + 16);   // K = 16+hi*8 .. 16+hi*8+7
  return concat8(lo, hs);
}

__device__ __forceinline__ v16h load_b_frag_t(const _Float16* BsT, int ld,
                                              int col0) {
  const int lane = threadIdx.x & 31;
  const int lid  = lane & 15;
  const int hi   = lane >> 4;
  const _Float16* r = BsT + (col0 + lid) * ld + (hi << 3);
  v8h lo = *(const v8h*)(r);        // K = hi*8 .. hi*8+7
  v8h hs = *(const v8h*)(r + 16);   // K = 16+hi*8 .. 16+hi*8+7
  return concat8(lo, hs);
}

__device__ __forceinline__ v8f wmma_f16(v16h a, v16h b, v8f c) {
  return __builtin_amdgcn_wmma_f32_16x16x32_f16(false, a, false, b, (short)0, c,
                                                false, false);
}

__device__ __forceinline__ float gelu_exact(float x) {
  return 0.5f * x * (1.0f + erff(x * 0.70710678118654752f));
}
__device__ __forceinline__ float silu(float x) {
  return x / (1.0f + expf(-x));
}

// ---------------------------------------------------------------------------
// Generic WMMA GEMM: C[M,N] = epilogue(A[M,K] @ W[K,N] + bias) (+resid)
// Block = 128 threads (4 waves). Tile 64 x BN, K-step 32. ACT: 0=none 1=gelu
// ---------------------------------------------------------------------------
template <int BN, int ACT>
__global__ __launch_bounds__(128) void k_gemm(
    const float* __restrict__ A, const float* __restrict__ W,
    const float* __restrict__ bias, const float* __restrict__ resid,
    float* __restrict__ C, int M, int Kd, int Nd) {
  constexpr int NS = BN / 16;
  __shared__ __align__(16) _Float16 As[64][40];
  __shared__ __align__(16) _Float16 BsT[BN][40];  // [col][k] transposed

  const int tid  = threadIdx.x;
  const int w    = tid >> 5;
  const int lane = tid & 31;
  const int lid  = lane & 15;
  const int hi   = lane >> 4;
  const int bm   = blockIdx.x * 64;
  const int bn   = blockIdx.y * BN;

  v8f acc[NS];
#pragma unroll
  for (int s = 0; s < NS; ++s)
#pragma unroll
    for (int j = 0; j < 8; ++j) acc[s][j] = 0.0f;

  for (int k0 = 0; k0 < Kd; k0 += 32) {
    // --- A tile: float4 global loads -> v4h LDS stores (row-major) ---
#pragma unroll
    for (int i = tid; i < 64 * 8; i += 128) {
      int r = i >> 3, c4 = (i & 7) << 2;
      int gr = bm + r, gk = k0 + c4;
      v4h h4;
      if (gr < M && gk + 3 < Kd) {
        const float4 f = *(const float4*)(A + (size_t)gr * Kd + gk);
        h4[0] = (_Float16)f.x; h4[1] = (_Float16)f.y;
        h4[2] = (_Float16)f.z; h4[3] = (_Float16)f.w;
      } else {
#pragma unroll
        for (int q = 0; q < 4; ++q)
          h4[q] = (gr < M && gk + q < Kd) ? (_Float16)A[(size_t)gr * Kd + gk + q]
                                          : (_Float16)0.f;
      }
      *(v4h*)(&As[r][c4]) = h4;
    }
    // --- B tile: float4 global loads -> transposed b16 LDS stores ---
    for (int i = tid; i < 32 * (BN / 4); i += 128) {
      int r = i / (BN / 4), c4 = (i % (BN / 4)) << 2;
      int gk = k0 + r, gc = bn + c4;
      _Float16 e0, e1, e2, e3;
      if (gk < Kd && gc + 3 < Nd && !(Nd & 3)) {
        const float4 f = *(const float4*)(W + (size_t)gk * Nd + gc);
        e0 = (_Float16)f.x; e1 = (_Float16)f.y;
        e2 = (_Float16)f.z; e3 = (_Float16)f.w;
      } else {
        e0 = (gk < Kd && gc + 0 < Nd) ? (_Float16)W[(size_t)gk * Nd + gc + 0] : (_Float16)0.f;
        e1 = (gk < Kd && gc + 1 < Nd) ? (_Float16)W[(size_t)gk * Nd + gc + 1] : (_Float16)0.f;
        e2 = (gk < Kd && gc + 2 < Nd) ? (_Float16)W[(size_t)gk * Nd + gc + 2] : (_Float16)0.f;
        e3 = (gk < Kd && gc + 3 < Nd) ? (_Float16)W[(size_t)gk * Nd + gc + 3] : (_Float16)0.f;
      }
      BsT[c4 + 0][r] = e0;
      BsT[c4 + 1][r] = e1;
      BsT[c4 + 2][r] = e2;
      BsT[c4 + 3][r] = e3;
    }
    // prefetch next A chunk (global_prefetch_b8)
    if (k0 + 32 < Kd) {
      int gr = bm + (tid & 63);
      if (gr < M) __builtin_prefetch(&A[(size_t)gr * Kd + k0 + 32], 0, 1);
    }
    __syncthreads();

    v16h a = load_a_frag(&As[w * 16][0], 40);
#pragma unroll
    for (int s = 0; s < NS; ++s) {
      v16h b = load_b_frag_t(&BsT[0][0], 40, s * 16);
      acc[s] = wmma_f16(a, b, acc[s]);
    }
    __syncthreads();
  }

#pragma unroll
  for (int s = 0; s < NS; ++s) {
#pragma unroll
    for (int i = 0; i < 8; ++i) {
      int row = bm + w * 16 + i + hi * 8;
      int col = bn + s * 16 + lid;
      if (row < M && col < Nd) {
        float v = acc[s][i];
        if (bias) v += bias[col];
        if (ACT == 1) v = gelu_exact(v);
        if (resid) v += resid[(size_t)row * Nd + col];
        C[(size_t)row * Nd + col] = v;
      }
    }
  }
}

// ---------------------------------------------------------------------------
// Row LayerNorm (in place), one wave per row.
// ---------------------------------------------------------------------------
__global__ __launch_bounds__(256) void k_layernorm(float* __restrict__ x,
                                                   const float* __restrict__ g,
                                                   const float* __restrict__ b,
                                                   int rows, int cols) {
  int wid  = threadIdx.x >> 5;
  int lane = threadIdx.x & 31;
  int row  = blockIdx.x * 8 + wid;
  if (row >= rows) return;
  float* r = x + (size_t)row * cols;
  float s = 0.f, ss = 0.f;
  for (int c = lane; c < cols; c += 32) {
    float v = r[c];
    s += v; ss += v * v;
  }
#pragma unroll
  for (int o = 16; o > 0; o >>= 1) {
    s  += __shfl_xor(s, o, 32);
    ss += __shfl_xor(ss, o, 32);
  }
  float mu  = s / cols;
  float var = ss / cols - mu * mu;
  float rs  = rsqrtf(var + 1e-5f);
  for (int c = lane; c < cols; c += 32) r[c] = (r[c] - mu) * rs * g[c] + b[c];
}

// ---------------------------------------------------------------------------
// Flash attention: one wave handles 16 queries of one (b, head).
// Scores via WMMA Q(16x32)·K^T(32x64); online softmax; P·V via WMMA.
// Ks  [key][dh] : B-operand layout for Q·K^T (col = key)
// VsT [dh][key] : B-operand layout for P·V   (col = dh)
// ---------------------------------------------------------------------------
__global__ __launch_bounds__(32) void k_attn(const float* __restrict__ Q,
                                             const float* __restrict__ Kv,
                                             const float* __restrict__ Vv,
                                             const unsigned char* __restrict__ nm,
                                             float* __restrict__ O) {
  const int b   = blockIdx.z;
  const int h   = blockIdx.y;
  const int q0  = blockIdx.x * 16;
  const int tid = threadIdx.x;
  const int lid = tid & 15;
  const int hi  = tid >> 4;
  const float scale = 0.20412414523193150f;  // 1/sqrt(24)

  __shared__ __align__(16) _Float16 Qs[16][40];
  __shared__ __align__(16) _Float16 Ks[64][40];
  __shared__ __align__(16) _Float16 VsT[32][72];
  __shared__ __align__(16) _Float16 Ps[16][72];
  __shared__ float amask[64];  // additive mask: 0 or -1e9

  // Q tile (float4 -> v4h), dh padded to 32 with zeros
#pragma unroll
  for (int i = tid; i < 16 * 8; i += 32) {
    int r = i >> 3, c4 = (i & 7) << 2;
    v4h h4;
    if (c4 < DH_) {
      const float4 f =
          *(const float4*)(Q + ((size_t)(b * N_ + q0 + r)) * H_ + h * DH_ + c4);
      h4[0] = (_Float16)f.x; h4[1] = (_Float16)f.y;
      h4[2] = (_Float16)f.z; h4[3] = (_Float16)f.w;
    } else {
      h4[0] = h4[1] = h4[2] = h4[3] = (_Float16)0.f;
    }
    *(v4h*)(&Qs[r][c4]) = h4;
  }
  __syncthreads();
  v16h qf = load_a_frag(&Qs[0][0], 40);

  float rm[8], rl[8];
  v8f acco[2];
#pragma unroll
  for (int i = 0; i < 8; ++i) { rm[i] = -1e30f; rl[i] = 0.f; }
#pragma unroll
  for (int s = 0; s < 2; ++s)
#pragma unroll
    for (int i = 0; i < 8; ++i) acco[s][i] = 0.f;

  for (int kb = 0; kb < N_; kb += 64) {
    for (int kk = tid; kk < 64; kk += 32) {
      int key = kb + kk;
      const float* kp = Kv + ((size_t)(b * N_ + key)) * H_ + h * DH_;
      const float* vp = Vv + ((size_t)(b * N_ + key)) * H_ + h * DH_;
#pragma unroll
      for (int d4 = 0; d4 < DH_; d4 += 4) {
        const float4 fk = *(const float4*)(kp + d4);
        v4h k4;
        k4[0] = (_Float16)fk.x; k4[1] = (_Float16)fk.y;
        k4[2] = (_Float16)fk.z; k4[3] = (_Float16)fk.w;
        *(v4h*)(&Ks[kk][d4]) = k4;
        const float4 fv = *(const float4*)(vp + d4);
        VsT[d4 + 0][kk] = (_Float16)fv.x;
        VsT[d4 + 1][kk] = (_Float16)fv.y;
        VsT[d4 + 2][kk] = (_Float16)fv.z;
        VsT[d4 + 3][kk] = (_Float16)fv.w;
      }
      {
        v4h z; z[0] = z[1] = z[2] = z[3] = (_Float16)0.f;
        *(v4h*)(&Ks[kk][24]) = z;
        *(v4h*)(&Ks[kk][28]) = z;
      }
#pragma unroll
      for (int d = DH_; d < 32; ++d) VsT[d][kk] = (_Float16)0.f;
      amask[kk] = nm[b * N_ + key] ? 0.f : -1e9f;
    }
    __syncthreads();

    v8f sa[4];
#pragma unroll
    for (int ns = 0; ns < 4; ++ns) {
      v16h bf = load_b_frag_t(&Ks[0][0], 40, ns * 16);
      v8f z;
#pragma unroll
      for (int j = 0; j < 8; ++j) z[j] = 0.f;
      sa[ns] = wmma_f16(qf, bf, z);
    }

#pragma unroll
    for (int i = 0; i < 8; ++i) {
      float sv[4];
      float mx = rm[i];
#pragma unroll
      for (int ns = 0; ns < 4; ++ns) {
        float v = sa[ns][i] * scale + amask[ns * 16 + lid];
        sv[ns] = v;
        mx = fmaxf(mx, v);
      }
#pragma unroll
      for (int o = 8; o > 0; o >>= 1) mx = fmaxf(mx, __shfl_xor(mx, o, 32));
      float alpha = expf(rm[i] - mx);
      float sum = 0.f;
#pragma unroll
      for (int ns = 0; ns < 4; ++ns) {
        float pe = expf(sv[ns] - mx);
        Ps[i + hi * 8][ns * 16 + lid] = (_Float16)pe;
        sum += pe;
      }
#pragma unroll
      for (int o = 8; o > 0; o >>= 1) sum += __shfl_xor(sum, o, 32);
      rl[i] = rl[i] * alpha + sum;
      rm[i] = mx;
      acco[0][i] *= alpha;
      acco[1][i] *= alpha;
    }
    __syncthreads();

#pragma unroll
    for (int kc = 0; kc < 64; kc += 32) {
      v16h pa = load_a_frag(&Ps[0][kc], 72);
#pragma unroll
      for (int ns2 = 0; ns2 < 2; ++ns2) {
        v16h vb = load_b_frag_t(&VsT[0][kc], 72, ns2 * 16);
        acco[ns2] = wmma_f16(pa, vb, acco[ns2]);
      }
    }
    __syncthreads();
  }

#pragma unroll
  for (int ns2 = 0; ns2 < 2; ++ns2) {
#pragma unroll
    for (int i = 0; i < 8; ++i) {
      int col = ns2 * 16 + lid;
      if (col < DH_) {
        int row = q0 + i + hi * 8;
        O[((size_t)(b * N_ + row)) * H_ + h * DH_ + col] = acco[ns2][i] / rl[i];
      }
    }
  }
}

// ---------------------------------------------------------------------------
__global__ void k_add4(const float4* __restrict__ a, const float4* __restrict__ b,
                       float4* __restrict__ c, size_t n4) {
  size_t i = (size_t)blockIdx.x * blockDim.x + threadIdx.x;
  if (i < n4) {
    float4 x = a[i], y = b[i];
    float4 r; r.x = x.x + y.x; r.y = x.y + y.y; r.z = x.z + y.z; r.w = x.w + y.w;
    c[i] = r;
  }
}

// ---------------------------------------------------------------------------
// Conditioning: timestep embedding -> c -> adaLN (sh,sc) for aa & ed. 1 block.
// ---------------------------------------------------------------------------
__global__ __launch_bounds__(384) void k_cond(
    const float* __restrict__ t,
    const float* __restrict__ tw1, const float* __restrict__ tb1,
    const float* __restrict__ tw2, const float* __restrict__ tb2,
    const float* __restrict__ aw1, const float* __restrict__ ab1,
    const float* __restrict__ aw2, const float* __restrict__ ab2,
    const float* __restrict__ ew1, const float* __restrict__ eb1,
    const float* __restrict__ ew2, const float* __restrict__ eb2,
    float* __restrict__ out_aa, float* __restrict__ out_ed) {
  __shared__ float tm[B_][FREQ_];
  __shared__ float h1[B_][H_];
  __shared__ float cb[B_][H_];
  const int tid = threadIdx.x;
  const float lg = logf(10000.f) / 128.f;

  if (tid < FREQ_) {
    for (int b = 0; b < B_; ++b) {
      if (tid < 128) tm[b][tid] = cosf(t[b] * expf(-lg * tid));
      else           tm[b][tid] = sinf(t[b] * expf(-lg * (tid - 128)));
    }
  }
  __syncthreads();
  {
    int j = tid;
    for (int b = 0; b < B_; ++b) {
      float acc = tb1[j];
      for (int i = 0; i < FREQ_; ++i) acc += tm[b][i] * tw1[i * H_ + j];
      h1[b][j] = silu(acc);
    }
  }
  __syncthreads();
  {
    int j = tid;
    for (int b = 0; b < B_; ++b) {
      float acc = tb2[j];
      for (int i = 0; i < H_; ++i) acc += h1[b][i] * tw2[i * H_ + j];
      cb[b][j] = acc;
    }
  }
  __syncthreads();
  {
    int j = tid;
    for (int b = 0; b < B_; ++b) {
      float acc = ab1[j];
      for (int i = 0; i < H_; ++i) acc += cb[b][i] * aw1[i * H_ + j];
      h1[b][j] = silu(acc);
    }
  }
  __syncthreads();
  if (tid < B_ * 2 * XD_) {
    int b = tid / (2 * XD_), m = tid % (2 * XD_);
    float acc = ab2[m];
    for (int i = 0; i < H_; ++i) acc += h1[b][i] * aw2[i * (2 * XD_) + m];
    out_aa[b * (2 * XD_) + m] = acc;
  }
  __syncthreads();
  {
    int j = tid;
    for (int b = 0; b < B_; ++b) {
      float acc = eb1[j];
      for (int i = 0; i < H_; ++i) acc += cb[b][i] * ew1[i * H_ + j];
      h1[b][j] = silu(acc);
    }
  }
  __syncthreads();
  if (tid < B_ * 2 * ED_) {
    int b = tid / (2 * ED_), m = tid % (2 * ED_);
    float acc = eb2[m];
    for (int i = 0; i < H_; ++i) acc += h1[b][i] * ew2[i * (2 * ED_) + m];
    out_ed[b * (2 * ED_) + m] = acc;
  }
}

// ---------------------------------------------------------------------------
// X = (x_in + (LN(aa) * (1+sc) + sh)) * mask, one thread per row.
// ---------------------------------------------------------------------------
__global__ void k_adaln_x(const float* __restrict__ aa,
                          const float* __restrict__ x_in,
                          const float* __restrict__ shsc,
                          const unsigned char* __restrict__ nm,
                          float* __restrict__ X) {
  int idx = blockIdx.x * blockDim.x + threadIdx.x;
  if (idx >= B_ * N_) return;
  int b = idx / N_;
  const float* a = aa + (size_t)idx * XD_;
  float s = 0.f, ss = 0.f;
#pragma unroll
  for (int c = 0; c < XD_; ++c) { s += a[c]; ss += a[c] * a[c]; }
  float mu = s / XD_, var = ss / XD_ - mu * mu;
  float rs = rsqrtf(var + 1e-5f);
  float mf = nm[idx] ? 1.f : 0.f;
#pragma unroll
  for (int c = 0; c < XD_; ++c) {
    float y = (a[c] - mu) * rs;
    y = y * (1.f + shsc[b * 2 * XD_ + XD_ + c]) + shsc[b * 2 * XD_ + c];
    X[(size_t)idx * XD_ + c] = (x_in[(size_t)idx * XD_ + c] + y) * mf;
  }
}

__global__ void k_adaln_e(const float* __restrict__ ev,
                          const float* __restrict__ shsc,
                          float* __restrict__ evf) {
  int idx = blockIdx.x * blockDim.x + threadIdx.x;
  if (idx >= B_ * N_) return;
  int b = idx / N_;
  const float* a = ev + (size_t)idx * ED_;
  float s = 0.f, ss = 0.f;
#pragma unroll
  for (int c = 0; c < ED_; ++c) { s += a[c]; ss += a[c] * a[c]; }
  float mu = s / ED_, var = ss / ED_ - mu * mu;
  float rs = rsqrtf(var + 1e-5f);
#pragma unroll
  for (int c = 0; c < ED_; ++c) {
    float y = (a[c] - mu) * rs;
    evf[(size_t)idx * ED_ + c] =
        y * (1.f + shsc[b * 2 * ED_ + ED_ + c]) + shsc[b * 2 * ED_ + c];
  }
}

// ---------------------------------------------------------------------------
__global__ void k_zero4(float4* __restrict__ p, size_t n4) {
  size_t i  = (size_t)blockIdx.x * blockDim.x + threadIdx.x;
  size_t st = (size_t)gridDim.x * blockDim.x;
  float4 z; z.x = z.y = z.z = z.w = 0.f;
  for (; i < n4; i += st) p[i] = z;
}

__global__ void k_scatter(const float* __restrict__ e,
                          const int* __restrict__ ei,
                          const int* __restrict__ batch,
                          float* __restrict__ E) {
  int t = blockIdx.x * blockDim.x + threadIdx.x;
  if (t >= TE_) return;
  int src = ei[t];
  int dst = ei[TE_ + t];
  int pid = batch[src];
  int ls = src - pid * N_;
  int ld = dst - pid * N_;
  size_t base = (((size_t)pid * N_ + ls) * N_ + ld) * ED_;
#pragma unroll
  for (int c = 0; c < ED_; ++c) E[base + c] = e[(size_t)t * ED_ + c];
}

// In-place: E = mask_zero * (0.5*(e_in + e_in^T) + ev_i*ev_j) * mf_i*mf_j
__global__ void k_efinal(float* __restrict__ E, const float* __restrict__ evf,
                         const unsigned char* __restrict__ nm) {
  size_t t = (size_t)blockIdx.x * blockDim.x + threadIdx.x;
  if (t >= (size_t)B_ * N_ * N_) return;
  int j = (int)(t % N_);
  int i = (int)((t / N_) % N_);
  int b = (int)(t / ((size_t)N_ * N_));
  if (j < i) return;  // each thread owns the (i,j)/(j,i) pair
  bool mi = nm[b * N_ + i] != 0;
  bool mj = nm[b * N_ + j] != 0;
  float mf   = (mi ? 1.f : 0.f) * (mj ? 1.f : 0.f);
  bool zero  = ((!mi) && (!mj)) || (i == j);
  size_t bij = (((size_t)b * N_ + i) * N_ + j) * ED_;
  size_t bji = (((size_t)b * N_ + j) * N_ + i) * ED_;
  const float* evi = evf + ((size_t)b * N_ + i) * ED_;
  const float* evj = evf + ((size_t)b * N_ + j) * ED_;
#pragma unroll
  for (int c = 0; c < ED_; ++c) {
    float eij = E[bij + c], eji = E[bji + c];
    float v = zero ? 0.f : (0.5f * (eij + eji) + evi[c] * evj[c]);
    v *= mf;
    E[bij + c] = v;
    E[bji + c] = v;
  }
}

// ---------------------------------------------------------------------------
extern "C" void kernel_launch(void* const* d_in, const int* in_sizes, int n_in,
                              void* d_out, int out_size, void* d_ws,
                              size_t ws_size, hipStream_t stream) {
  const float* W_node = (const float*)d_in[0];
  const float* wq  = (const float*)d_in[1];
  const float* wk  = (const float*)d_in[2];
  const float* wv  = (const float*)d_in[3];
  const float* wo  = (const float*)d_in[4];
  const float* tw1 = (const float*)d_in[5];
  const float* tw2 = (const float*)d_in[6];
  const float* aaw1 = (const float*)d_in[7];
  const float* aaw2 = (const float*)d_in[8];
  const float* edw1 = (const float*)d_in[9];
  const float* edw2 = (const float*)d_in[10];
  const float* adaw1 = (const float*)d_in[11];
  const float* adaw2 = (const float*)d_in[12];
  const float* adew1 = (const float*)d_in[13];
  const float* adew2 = (const float*)d_in[14];
  const float* bq  = (const float*)d_in[15];
  const float* bk  = (const float*)d_in[16];
  const float* bv  = (const float*)d_in[17];
  const float* bo  = (const float*)d_in[18];
  const float* tb1 = (const float*)d_in[19];
  const float* tb2 = (const float*)d_in[20];
  const float* aab1 = (const float*)d_in[21];
  const float* aab2 = (const float*)d_in[22];
  const float* edb1 = (const float*)d_in[23];
  const float* edb2 = (const float*)d_in[24];
  const float* adab1 = (const float*)d_in[25];
  const float* adab2 = (const float*)d_in[26];
  const float* adeb1 = (const float*)d_in[27];
  const float* adeb2 = (const float*)d_in[28];
  const float* ln1g = (const float*)d_in[29];
  const float* ln1b = (const float*)d_in[30];
  const float* ln2g = (const float*)d_in[31];
  const float* ln2b = (const float*)d_in[32];
  const float* x  = (const float*)d_in[33];
  const float* e  = (const float*)d_in[34];
  const float* p  = (const float*)d_in[35];
  const unsigned char* nm = (const unsigned char*)d_in[36];
  const float* t  = (const float*)d_in[38];
  const int* ei   = (const int*)d_in[39];
  const int* batch = (const int*)d_in[40];

  float* ws = (float*)d_ws;
  const size_t SZ = (size_t)B_ * N_ * H_;
  float* S0  = ws;
  float* P0  = ws + 1 * SZ;
  float* Qb  = ws + 2 * SZ;
  float* Kb  = ws + 3 * SZ;
  float* Vb  = ws + 4 * SZ;
  float* Ob  = ws + 5 * SZ;
  float* Hb  = ws + 6 * SZ;
  float* HID = Qb;  // reuse Q..O (exactly 4*SZ = B*N*MLPH) after attention done
  float* AA  = ws + 7 * SZ;
  float* EVR = AA + (size_t)B_ * N_ * XD_;
  float* EVF = EVR + (size_t)B_ * N_ * ED_;
  float* CAA = EVF + (size_t)B_ * N_ * ED_;
  float* CED = CAA + B_ * 2 * XD_;

  float* Xout = (float*)d_out;
  float* Eout = Xout + (size_t)B_ * N_ * XD_;

  const int M = B_ * N_;
  dim3 gP(128, 3);   // M/64 x H/128
  dim3 gH(128, 12);  // M/64 x MLPH/128
  dim3 gS(128, 1);   // small-N

  // struct/seq embeddings + LN
  k_gemm<128, 0><<<gP, 128, 0, stream>>>(x, W_node, nullptr, nullptr, S0, M, XD_, H_);
  k_gemm<128, 0><<<gP, 128, 0, stream>>>(p, W_node, nullptr, nullptr, P0, M, XD_, H_);
  k_layernorm<<<M / 8, 256, 0, stream>>>(S0, ln1g, ln1b, M, H_);
  k_layernorm<<<M / 8, 256, 0, stream>>>(P0, ln2g, ln2b, M, H_);

  // MHA1: struct attends to seq
  k_gemm<128, 0><<<gP, 128, 0, stream>>>(S0, wq, bq, nullptr, Qb, M, H_, H_);
  k_gemm<128, 0><<<gP, 128, 0, stream>>>(P0, wk, bk, nullptr, Kb, M, H_, H_);
  k_gemm<128, 0><<<gP, 128, 0, stream>>>(P0, wv, bv, nullptr, Vb, M, H_, H_);
  k_attn<<<dim3(N_ / 16, NH_, B_), 32, 0, stream>>>(Qb, Kb, Vb, nm, Ob);
  k_gemm<128, 0><<<gP, 128, 0, stream>>>(Ob, wo, bo, S0, S0, M, H_, H_);  // struct1

  // MHA2: seq attends to UPDATED struct
  k_gemm<128, 0><<<gP, 128, 0, stream>>>(P0, wq, bq, nullptr, Qb, M, H_, H_);
  k_gemm<128, 0><<<gP, 128, 0, stream>>>(S0, wk, bk, nullptr, Kb, M, H_, H_);
  k_gemm<128, 0><<<gP, 128, 0, stream>>>(S0, wv, bv, nullptr, Vb, M, H_, H_);
  k_attn<<<dim3(N_ / 16, NH_, B_), 32, 0, stream>>>(Qb, Kb, Vb, nm, Ob);
  k_gemm<128, 0><<<gP, 128, 0, stream>>>(Ob, wo, bo, P0, P0, M, H_, H_);  // seq1

  // h = struct1 + seq1
  k_add4<<<(unsigned)((SZ / 4 + 255) / 256), 256, 0, stream>>>(
      (const float4*)S0, (const float4*)P0, (float4*)Hb, SZ / 4);

  // conditioning
  k_cond<<<1, 384, 0, stream>>>(t, tw1, tb1, tw2, tb2, adaw1, adab1, adaw2,
                                adab2, adew1, adeb1, adew2, adeb2, CAA, CED);

  // aa branch: gelu(h@aa_w1+b) @ aa_w2 + b -> adaLN -> X
  k_gemm<128, 1><<<gH, 128, 0, stream>>>(Hb, aaw1, aab1, nullptr, HID, M, H_, MLPH_);
  k_gemm<32, 0><<<gS, 128, 0, stream>>>(HID, aaw2, aab2, nullptr, AA, M, MLPH_, XD_);
  k_adaln_x<<<(M + 255) / 256, 256, 0, stream>>>(AA, x, CAA, nm, Xout);

  // ed branch: gelu(h@ed_w1+b) @ ed_w2 + b -> adaLN -> ev
  k_gemm<128, 1><<<gH, 128, 0, stream>>>(Hb, edw1, edb1, nullptr, HID, M, H_, MLPH_);
  k_gemm<32, 0><<<gS, 128, 0, stream>>>(HID, edw2, edb2, nullptr, EVR, M, MLPH_, ED_);
  k_adaln_e<<<(M + 255) / 256, 256, 0, stream>>>(EVR, CED, EVF);

  // E assembly (in d_out; no 168MB scratch)
  size_t En = (size_t)B_ * N_ * N_ * ED_;
  k_zero4<<<8192, 256, 0, stream>>>((float4*)Eout, En / 4);
  k_scatter<<<(TE_ + 255) / 256, 256, 0, stream>>>(e, ei, batch, Eout);
  size_t pairs = (size_t)B_ * N_ * N_;
  k_efinal<<<(unsigned)((pairs + 255) / 256), 256, 0, stream>>>(Eout, EVF, nm);
}
